// EquiMessagePassingLayer_89653147337401
// MI455X (gfx1250) — compile-verified, matrix operand order checked by reading, not run on hardware
//
#include <hip/hip_runtime.h>
#include <math.h>

#define B_   4
#define S_   64
#define N_   256
#define D_   384
#define DM_  64
#define H_   32
#define P_   64
#define EPS_ 1e-6f

typedef _Float16 h8   __attribute__((ext_vector_type(8)));
typedef _Float16 v16h __attribute__((ext_vector_type(16)));
typedef float    v8f  __attribute__((ext_vector_type(8)));

// Load one 16x32 (A) or 32x16 (B) f16 WMMA fragment from LDS.
// Per CDNA5 ISA: lane L<16 -> row L, K = {kh*8+0..7, 16+kh*8+0..7} with kh=L>>4.
__device__ __forceinline__ v16h frag_ld(const _Float16* base, int stride, int row,
                                        int khalf, int k0) {
  union { v16h v; h8 h[2]; } u;
  const _Float16* rp = base + row * stride + k0 + khalf * 8;
  u.h[0] = *(const h8*)(rp);
  u.h[1] = *(const h8*)(rp + 16);
  return u.v;
}

// ---------------------------------------------------------------------------
// K1: coordinate centering (em path, 'none') + length-normalized (ca path).
// Writes node-major transposed layouts cT/cnT[(b*N+i)*S*3 + s*3 + d].
// ---------------------------------------------------------------------------
__global__ void k_center(const float* __restrict__ coords,
                         const float* __restrict__ node_mask,
                         const float* __restrict__ atom_mask,
                         const float* __restrict__ em_sw,
                         const float* __restrict__ ca_sw,
                         float* __restrict__ cT, float* __restrict__ cnT) {
  __shared__ float red[256];
  int bs = blockIdx.x;           // b*S + s
  int s  = bs % S_;
  int b  = bs / S_;
  int i  = threadIdx.x;
  size_t mi = (size_t)bs * N_ + i;
  float x0 = coords[mi * 3 + 0], x1 = coords[mi * 3 + 1], x2 = coords[mi * 3 + 2];
  float am = atom_mask[mi], nm = node_mask[mi];

  float sums[4];
  float vals[4] = {x0 * am, x1 * am, x2 * am, am};
  for (int r = 0; r < 4; ++r) {
    red[i] = vals[r]; __syncthreads();
    for (int st = 128; st > 0; st >>= 1) { if (i < st) red[i] += red[i + st]; __syncthreads(); }
    sums[r] = red[0]; __syncthreads();
  }
  float invc = 1.f / sums[3];
  float c0 = (x0 - sums[0] * invc) * nm;
  float c1 = (x1 - sums[1] * invc) * nm;
  float c2 = (x2 - sums[2] * invc) * nm;

  size_t ti = ((size_t)(b * N_ + i) * S_ + s) * 3;
  float ews = em_sw[s] * nm;
  cT[ti + 0] = c0 * ews; cT[ti + 1] = c1 * ews; cT[ti + 2] = c2 * ews;

  float sq  = c0 * c0 + c1 * c1 + c2 * c2;
  float len = sqrtf(sq);
  red[i] = len; __syncthreads();
  for (int st = 128; st > 0; st >>= 1) { if (i < st) red[i] += red[i + st]; __syncthreads(); }
  float slen = red[0]; __syncthreads();
  red[i] = nm; __syncthreads();
  for (int st = 128; st > 0; st >>= 1) { if (i < st) red[i] += red[i + st]; __syncthreads(); }
  float snm = red[0]; __syncthreads();

  float div = slen / snm + EPS_;
  float cws = ca_sw[s] / div * nm;
  cnT[ti + 0] = c0 * cws; cnT[ti + 1] = c1 * cws; cnT[ti + 2] = c2 * cws;
}

// ---------------------------------------------------------------------------
// K2: per-node dual LayerNorm, p = LN_em @ em_proj_W^T + b, q = LN_na @ na_in_W^T + b
// ---------------------------------------------------------------------------
__global__ void k_node_prep(const float* __restrict__ nfeat,
                            const float* __restrict__ em_g, const float* __restrict__ em_b,
                            const float* __restrict__ na_g, const float* __restrict__ na_b,
                            const float* __restrict__ pW, const float* __restrict__ pB,
                            const float* __restrict__ qW, const float* __restrict__ qB,
                            float* __restrict__ wsP, float* __restrict__ wsQ) {
  __shared__ float xs[D_], nf1[D_], nf2[D_], red[256];
  int r = blockIdx.x, t = threadIdx.x;
  const float* row = nfeat + (size_t)r * D_;
  for (int k = t; k < D_; k += 256) xs[k] = row[k];
  __syncthreads();

  float a = xs[t] + ((t + 256 < D_) ? xs[t + 256] : 0.f);
  red[t] = a; __syncthreads();
  for (int st = 128; st > 0; st >>= 1) { if (t < st) red[t] += red[t + st]; __syncthreads(); }
  float mu = red[0] / (float)D_; __syncthreads();
  float d0 = xs[t] - mu;
  float d1 = (t + 256 < D_) ? (xs[t + 256] - mu) : 0.f;
  red[t] = d0 * d0 + d1 * d1; __syncthreads();
  for (int st = 128; st > 0; st >>= 1) { if (t < st) red[t] += red[t + st]; __syncthreads(); }
  float rs = rsqrtf(red[0] / (float)D_ + 1e-5f); __syncthreads();

  for (int k = t; k < D_; k += 256) {
    float nv = (xs[k] - mu) * rs;
    nf1[k] = nv * em_g[k] + em_b[k];
    nf2[k] = nv * na_g[k] + na_b[k];
  }
  __syncthreads();

  if (t < DM_) {
    const float* w = pW + (size_t)t * D_;
    float acc = pB[t];
    for (int k = 0; k < D_; ++k) acc += nf1[k] * w[k];
    wsP[(size_t)r * DM_ + t] = acc;
  }
  for (int o = t; o < D_; o += 256) {
    const float* w = qW + (size_t)o * D_;
    float acc = qB[o];
    for (int k = 0; k < D_; ++k) acc += nf2[k] * w[k];
    wsQ[(size_t)r * D_ + o] = acc;
  }
}

// ---------------------------------------------------------------------------
// K3: per-node e_start = b1 + p@W1[:, :64]^T, e_end = p@W1[:, 64:128]^T,
//     proj[b,d,n,p] = sum_s cn[b,s,n,d] * ca_proj_W[p,s]
// ---------------------------------------------------------------------------
__global__ void k_edge_prep(const float* __restrict__ wsP, const float* __restrict__ cnT,
                            const float* __restrict__ W1, const float* __restrict__ b1,
                            const float* __restrict__ projW,
                            float* __restrict__ wsES, float* __restrict__ wsEE,
                            float* __restrict__ wsProj) {
  __shared__ float pL[DM_], cL[S_ * 3];
  int r = blockIdx.x, t = threadIdx.x;
  int b = r / N_, n = r % N_;
  if (t < DM_) pL[t] = wsP[(size_t)r * DM_ + t];
  if (t < S_ * 3) cL[t] = cnT[(size_t)r * S_ * 3 + t];
  __syncthreads();
  for (int o = t; o < 384; o += 256) {
    if (o < 96) {
      const float* w = W1 + (size_t)o * 192;
      float acc = b1[o];
      for (int k = 0; k < DM_; ++k) acc += pL[k] * w[k];
      wsES[(size_t)r * 96 + o] = acc;
    } else if (o < 192) {
      int f = o - 96;
      const float* w = W1 + (size_t)f * 192 + 64;
      float acc = 0.f;
      for (int k = 0; k < DM_; ++k) acc += pL[k] * w[k];
      wsEE[(size_t)r * 96 + f] = acc;
    } else {
      int u = o - 192, d = u / P_, pp = u % P_;
      const float* w = projW + (size_t)pp * S_;
      float acc = 0.f;
      for (int s = 0; s < S_; ++s) acc += cL[s * 3 + d] * w[s];
      wsProj[((size_t)(b * 3 + d) * N_ + n) * P_ + pp] = acc;
    }
  }
}

// ---------------------------------------------------------------------------
// K4: fully fused per (b, i): cf gram -> WMMA edge MLP -> dual softmax ->
//     node attention + na_out, coord attention + ca_out.  Messages live only
//     in LDS (256 j x 96 f x f32 = 96KB, enabled by 320KB LDS/WGP).
//     Per K-step the 6 B fragments are batch-loaded (48 VGPRs) so the 6-WMMA
//     chain issues behind a single dscnt wait, without spill-inducing hoists.
// ---------------------------------------------------------------------------
#define F32_CNT (256*96 + 768 + 384 + 192 + 192 + 192 + 96 + 256 + 256 + 96 + 96 + 96)
#define F16_CNT (96*64 + 96*96 + 8*16*64 + 8*16*96)
#define FUSED_SMEM ((size_t)F32_CNT * 4 + (size_t)F16_CNT * 2)

__global__ void k_fused(const float* __restrict__ cT,  const float* __restrict__ esG,
                        const float* __restrict__ eeG, const float* __restrict__ qG,
                        const float* __restrict__ projG, const int* __restrict__ adj,
                        const float* __restrict__ W1, const float* __restrict__ W2,
                        const float* __restrict__ b2,
                        const float* __restrict__ naOutW, const float* __restrict__ naOutB,
                        const float* __restrict__ caOutW,
                        const float* __restrict__ nfeat, const float* __restrict__ coords,
                        float* __restrict__ outC, float* __restrict__ outN) {
  extern __shared__ char smem[];
  float* msgL  = (float*)smem;          // 256*96
  float* redF  = msgL + 256 * 96;       // 4*3*64
  float* outn  = redF + 768;            // 384
  float* updL  = outn + 384;            // 192
  float* ciT   = updL + 192;            // 192
  float* projI = ciT + 192;             // 192
  float* esL   = projI + 192;           // 96
  float* maskv = esL + 96;              // 256
  float* red   = maskv + 256;           // 256
  float* smax  = red + 256;             // 96
  float* ssum  = smax + 96;             // 96
  float* swv   = ssum + 96;             // 96
  _Float16* Bh1 = (_Float16*)(swv + 96);   // 96x64  (W1c, [f][s])
  _Float16* Bh2 = Bh1 + 96 * 64;           // 96x96  (W2,  [f][k])
  _Float16* cfT = Bh2 + 96 * 96;           // 8 waves x 16x64
  _Float16* hT  = cfT + 8 * 16 * 64;       // 8 waves x 16x96

  int blk = blockIdx.x;
  int b = blk / N_, i = blk % N_;
  int t = threadIdx.x;
  int lane = t & 31, w = t >> 5;
  int m16 = lane & 15, kh = lane >> 4;

  // ---- Phase A: stage weights (f32 -> f16), per-i rows, adjacency mask ----
  for (int idx = t; idx < 96 * 64; idx += 256)
    Bh1[idx] = (_Float16)W1[(size_t)(idx / 64) * 192 + 128 + (idx % 64)];
  for (int idx = t; idx < 96 * 96; idx += 256)
    Bh2[idx] = (_Float16)W2[idx];
  if (t < 192) ciT[t]   = cT[(size_t)blk * 192 + t];
  if (t < 192) projI[t] = projG[((size_t)(b * 3 + t / 64) * N_ + i) * 64 + (t % 64)];
  if (t < 96)  esL[t]   = esG[(size_t)blk * 96 + t];
  int av = adj[(size_t)blk * N_ + t];
  red[t] = av ? 1.f : 0.f;
  __syncthreads();
  for (int st = 128; st > 0; st >>= 1) { if (t < st) red[t] += red[t + st]; __syncthreads(); }
  float deg = red[0];
  maskv[t] = (deg == 0.f) ? 0.f : (av ? 0.f : -1e30f);
  __syncthreads();

  // ---- Phase B: per-wave edge tiles (16 j each), two WMMA GEMM layers ----
  _Float16* cfW = cfT + w * 16 * 64;
  _Float16* hW  = hT  + w * 16 * 96;
#pragma unroll 1
  for (int tile = 0; tile < 2; ++tile) {
    int jbase = (w + tile * 8) * 16;
    // cf[jj][s] = <c_em[b,s,i,:], c_em[b,s,j,:]> ; lane owns jj=m16, s-half=kh
    {
      int jj = m16;
      const float* cj = cT + (size_t)(b * N_ + jbase + jj) * S_ * 3;
      for (int s = kh * 32; s < kh * 32 + 32; ++s) {
        float v = ciT[s*3+0]*cj[s*3+0] + ciT[s*3+1]*cj[s*3+1] + ciT[s*3+2]*cj[s*3+2];
        cfW[jj * 64 + s] = (_Float16)v;
      }
    }
    // Layer 1: h = silu(e_start_i + e_end_j + cf @ W1c^T)   [K=64]
    v8f acc[6] = {};
#pragma unroll
    for (int kk = 0; kk < 2; ++kk) {
      v16h afr = frag_ld(cfW, 64, m16, kh, kk * 32);
      v16h bfr[6];
#pragma unroll
      for (int ft = 0; ft < 6; ++ft)
        bfr[ft] = frag_ld(Bh1, 64, ft * 16 + m16, kh, kk * 32);
#pragma unroll
      for (int ft = 0; ft < 6; ++ft)
        acc[ft] = __builtin_amdgcn_wmma_f32_16x16x32_f16(
            false, afr, false, bfr[ft], (short)0, acc[ft], false, false);
    }
#pragma unroll
    for (int ft = 0; ft < 6; ++ft) {
      int f = ft * 16 + m16;
      float esf = esL[f];
#pragma unroll
      for (int r = 0; r < 8; ++r) {
        int m = kh * 8 + r, j = jbase + m;
        float v = acc[ft][r] + esf + eeG[(size_t)(b * N_ + j) * 96 + f];
        v = v / (1.f + __expf(-v));          // silu
        hW[m * 96 + f] = (_Float16)v;
      }
    }
    // Layer 2: msg = h @ W2^T + b2   [K=96]
    v8f acc2[6] = {};
#pragma unroll
    for (int kk = 0; kk < 3; ++kk) {
      v16h afr = frag_ld(hW, 96, m16, kh, kk * 32);
      v16h bfr[6];
#pragma unroll
      for (int ft = 0; ft < 6; ++ft)
        bfr[ft] = frag_ld(Bh2, 96, ft * 16 + m16, kh, kk * 32);
#pragma unroll
      for (int ft = 0; ft < 6; ++ft)
        acc2[ft] = __builtin_amdgcn_wmma_f32_16x16x32_f16(
            false, afr, false, bfr[ft], (short)0, acc2[ft], false, false);
    }
#pragma unroll
    for (int ft = 0; ft < 6; ++ft) {
      int f = ft * 16 + m16;
      float bb = b2[f];
#pragma unroll
      for (int r = 0; r < 8; ++r) {
        int m = kh * 8 + r, j = jbase + m;
        msgL[j * 96 + f] = acc2[ft][r] + bb;
      }
    }
  }
  __syncthreads();

  // ---- Phase C: node softmax over j (32 heads) + w_n ----
  {
    int h = t & 31, ch = t >> 5;
    float lmax = -3.0e38f;
    for (int jj = 0; jj < 32; ++jj) {
      int j = ch * 32 + jj;
      lmax = fmaxf(lmax, msgL[j * 96 + h] + maskv[j]);
    }
    red[t] = lmax; __syncthreads();
    if (ch == 0) { float mm = red[h]; for (int c = 1; c < 8; ++c) mm = fmaxf(mm, red[c*32+h]); smax[h] = mm; }
    __syncthreads();
    float mm = smax[h], ls = 0.f;
    for (int jj = 0; jj < 32; ++jj) {
      int j = ch * 32 + jj;
      ls += __expf(msgL[j * 96 + h] + maskv[j] - mm);
    }
    red[t] = ls; __syncthreads();
    if (ch == 0) { float sm = 0.f; for (int c = 0; c < 8; ++c) sm += red[c*32+h]; ssum[h] = sm; }
    __syncthreads();
    float inv = 1.f / ssum[h], l2 = 0.f;
    for (int jj = 0; jj < 32; ++jj) {
      int j = ch * 32 + jj;
      float a = __expf(msgL[j * 96 + h] + maskv[j] - mm) * inv;
      msgL[j * 96 + h] = a;
      l2 += a * a;
    }
    red[t] = l2; __syncthreads();
    if (ch == 0) { float sm = 0.f; for (int c = 0; c < 8; ++c) sm += red[c*32+h]; swv[h] = sqrtf(sm); }
    __syncthreads();
  }

  // ---- Phase D: coord softmax over j (64 channels) + w_c ----
  {
    int p = t & 63, ch = t >> 6;
    float lmax = -3.0e38f;
    for (int jj = 0; jj < 64; ++jj) {
      int j = ch * 64 + jj;
      lmax = fmaxf(lmax, msgL[j * 96 + 32 + p] + maskv[j]);
    }
    red[t] = lmax; __syncthreads();
    if (ch == 0) { float mm = red[p]; for (int c = 1; c < 4; ++c) mm = fmaxf(mm, red[c*64+p]); smax[32+p] = mm; }
    __syncthreads();
    float mm = smax[32 + p], ls = 0.f;
    for (int jj = 0; jj < 64; ++jj) {
      int j = ch * 64 + jj;
      ls += __expf(msgL[j * 96 + 32 + p] + maskv[j] - mm);
    }
    red[t] = ls; __syncthreads();
    if (ch == 0) { float sm = 0.f; for (int c = 0; c < 4; ++c) sm += red[c*64+p]; ssum[32+p] = sm; }
    __syncthreads();
    float inv = 1.f / ssum[32 + p], l2 = 0.f;
    for (int jj = 0; jj < 64; ++jj) {
      int j = ch * 64 + jj;
      float a = __expf(msgL[j * 96 + 32 + p] + maskv[j] - mm) * inv;
      msgL[j * 96 + 32 + p] = a;
      l2 += a * a;
    }
    red[t] = l2; __syncthreads();
    if (ch == 0) { float sm = 0.f; for (int c = 0; c < 4; ++c) sm += red[c*64+p]; swv[32+p] = sqrtf(sm); }
    __syncthreads();
  }

  // ---- Phase E: node attention apply + na_out projection ----
  for (int o = t; o < D_; o += 256) {
    int h = o / 12;
    float a = 0.f;
    for (int j = 0; j < N_; ++j)
      a += msgL[j * 96 + h] * qG[(size_t)(b * N_ + j) * D_ + o];
    outn[o] = a * swv[h];
  }
  __syncthreads();
  for (int o = t; o < D_; o += 256) {
    const float* wr = naOutW + (size_t)o * D_;
    float a = naOutB[o];
    for (int k = 0; k < D_; ++k) a += outn[k] * wr[k];
    size_t idx = (size_t)blk * D_ + o;
    outN[idx] = nfeat[idx] + a;
  }

  // ---- Phase F: coord attention apply + ca_out projection ----
  {
    int p = t & 63, jc = t >> 6;
    float a0 = 0.f, a1 = 0.f, a2 = 0.f;
    float pi0 = projI[p], pi1 = projI[64 + p], pi2 = projI[128 + p];
    const float* pj0 = projG + (size_t)(b * 3 + 0) * N_ * 64;
    const float* pj1 = projG + (size_t)(b * 3 + 1) * N_ * 64;
    const float* pj2 = projG + (size_t)(b * 3 + 2) * N_ * 64;
    for (int jj = 0; jj < 64; ++jj) {
      int j = jc * 64 + jj;
      float at = msgL[j * 96 + 32 + p];
      float v0 = pi0 - pj0[(size_t)j * 64 + p];
      float v1 = pi1 - pj1[(size_t)j * 64 + p];
      float v2 = pi2 - pj2[(size_t)j * 64 + p];
      float sq = v0 * v0 + v1 * v1 + v2 * v2;
      float inv = 1.f / (sqrtf(sq) + EPS_);
      a0 += at * v0 * inv; a1 += at * v1 * inv; a2 += at * v2 * inv;
    }
    redF[(jc * 3 + 0) * 64 + p] = a0;
    redF[(jc * 3 + 1) * 64 + p] = a1;
    redF[(jc * 3 + 2) * 64 + p] = a2;
  }
  __syncthreads();
  if (t < 192) {
    int d = t / 64, pp = t % 64;
    float u = 0.f;
    for (int jc = 0; jc < 4; ++jc) u += redF[(jc * 3 + d) * 64 + pp];
    updL[d * 64 + pp] = u * swv[32 + pp];
  }
  __syncthreads();
  if (t < 192) {
    int s = t / 3, d = t % 3;
    const float* wr = caOutW + (size_t)s * 64;
    const float* ul = updL + d * 64;
    float a = 0.f;
    for (int pp = 0; pp < 64; ++pp) a += ul[pp] * wr[pp];
    size_t idx = ((size_t)(b * S_ + s) * N_ + i) * 3 + d;
    outC[idx] = coords[idx] + a;
  }
}

// ---------------------------------------------------------------------------
extern "C" void kernel_launch(void* const* d_in, const int* in_sizes, int n_in,
                              void* d_out, int out_size, void* d_ws, size_t ws_size,
                              hipStream_t stream) {
  (void)in_sizes; (void)n_in; (void)out_size; (void)ws_size;
  const float* coords     = (const float*)d_in[0];
  const float* node_feats = (const float*)d_in[1];
  const int*   adj        = (const int*)d_in[2];
  const float* node_mask  = (const float*)d_in[3];
  const float* atom_mask  = (const float*)d_in[4];
  const float* em_sw      = (const float*)d_in[5];
  const float* em_ln_g    = (const float*)d_in[6];
  const float* em_ln_b    = (const float*)d_in[7];
  const float* em_proj_W  = (const float*)d_in[8];
  const float* em_proj_b  = (const float*)d_in[9];
  const float* em_W1      = (const float*)d_in[10];
  const float* em_b1      = (const float*)d_in[11];
  const float* em_W2      = (const float*)d_in[12];
  const float* em_b2      = (const float*)d_in[13];
  const float* ca_sw      = (const float*)d_in[14];
  const float* ca_proj_W  = (const float*)d_in[15];
  const float* ca_out_W   = (const float*)d_in[16];
  const float* na_ln_g    = (const float*)d_in[17];
  const float* na_ln_b    = (const float*)d_in[18];
  const float* na_in_W    = (const float*)d_in[19];
  const float* na_in_b    = (const float*)d_in[20];
  const float* na_out_W   = (const float*)d_in[21];
  const float* na_out_b   = (const float*)d_in[22];

  float* ws     = (float*)d_ws;
  float* cT     = ws;                  // B*N*S*3 = 196608
  float* cnT    = cT + 196608;         // 196608
  float* wsP    = cnT + 196608;        // B*N*64  = 65536
  float* wsQ    = wsP + 65536;         // B*N*384 = 393216
  float* wsES   = wsQ + 393216;        // B*N*96  = 98304
  float* wsEE   = wsES + 98304;        // 98304
  float* wsProj = wsEE + 98304;        // B*3*N*64 = 196608

  float* outC = (float*)d_out;
  float* outN = outC + (size_t)B_ * S_ * N_ * 3;

  k_center<<<B_ * S_, 256, 0, stream>>>(coords, node_mask, atom_mask, em_sw, ca_sw, cT, cnT);
  k_node_prep<<<B_ * N_, 256, 0, stream>>>(node_feats, em_ln_g, em_ln_b, na_ln_g, na_ln_b,
                                           em_proj_W, em_proj_b, na_in_W, na_in_b, wsP, wsQ);
  k_edge_prep<<<B_ * N_, 256, 0, stream>>>(wsP, cnT, em_W1, em_b1, ca_proj_W,
                                           wsES, wsEE, wsProj);
  k_fused<<<B_ * N_, 256, FUSED_SMEM, stream>>>(cT, wsES, wsEE, wsQ, wsProj, adj,
                                                em_W1, em_W2, em_b2,
                                                na_out_W, na_out_b, ca_out_W,
                                                node_feats, coords, outC, outN);
}
